// GroupedQueryAttention_83923660964371
// MI455X (gfx1250) — matrix operand order repaired and checked by reading, not compiled, not run on hardware
//
#include <hip/hip_runtime.h>

#define T_SEQ 2048
#define HID 4096
#define NHEADS 32
#define NKV 8
#define HDIM 128

typedef __attribute__((ext_vector_type(16))) __bf16 v16bf;
typedef __attribute__((ext_vector_type(8)))  float  v8f;
typedef __attribute__((ext_vector_type(4)))  unsigned int v4u;
typedef __attribute__((ext_vector_type(8)))  unsigned int v8u;

union AFrag { v16bf v; unsigned int u[8]; };
union CFrag { v8f v; float f[8]; };

__device__ __forceinline__ unsigned short f2bf(float f) {
  unsigned int x = __float_as_uint(f);
  x += 0x7FFFu + ((x >> 16) & 1u);          // round-to-nearest-even
  return (unsigned short)(x >> 16);
}
__device__ __forceinline__ float bf2f(unsigned short h) {
  return __uint_as_float(((unsigned int)h) << 16);
}
__device__ __forceinline__ unsigned short ldbf(const float* p) { return f2bf(*p); }
__device__ __forceinline__ unsigned short ldbf(const unsigned short* p) { return *p; }
__device__ __forceinline__ void stf(float* p, float v) { *p = v; }
__device__ __forceinline__ void stf(unsigned short* p, float v) { *p = f2bf(v); }

// ---------------------------------------------------------------------------
// Generic GEMM: C[M,N] = A[M,K] * B[N,K]^T   (bf16 WMMA, f32 accumulate)
// Block tile 128(M) x 64(N), 8 waves (wave32), wave tile 32x32, K-step 32.
// When A is already bf16 in memory, the A tile is staged by the Tensor Data
// Mover (tensor_load_to_lds, 2-D descriptor, LDS padding 4B per 64B row).
// ---------------------------------------------------------------------------
template <typename TA, typename TB, typename TC>
__global__ __launch_bounds__(256)
void gemm_bf16_wmma(const TA* __restrict__ A, int lda,
                    const TB* __restrict__ B, int ldb,
                    TC* __restrict__ C, int ldc, int K) {
  constexpr bool kTdmA = __is_same(TA, unsigned short);

  const int gm0  = blockIdx.y * 128;
  const int gn0  = blockIdx.x * 64;
  const int tid  = threadIdx.x;
  const int lane = tid & 31;
  const int wave = tid >> 5;
  const int wm   = (wave >> 1) * 32;
  const int wn   = (wave & 1) * 32;
  const int lrow = lane & 15;
  const int lhi  = lane >> 4;
  const int kgrp = lhi * 8;

  __shared__ unsigned short As[128][34];  // [m][k], 64B row + 4B pad
  __shared__ unsigned short Bt[32][66];   // [k][n], padded (B staged transposed)

  CFrag c[2][2];
#pragma unroll
  for (int i = 0; i < 2; ++i)
#pragma unroll
    for (int j = 0; j < 2; ++j)
#pragma unroll
      for (int r = 0; r < 8; ++r) c[i][j].f[r] = 0.0f;

  const int am = tid >> 1;          // 0..127
  const int ak = (tid & 1) * 16;    // 0/16
  const int bn = tid >> 2;          // 0..63
  const int bk = (tid & 3) * 8;     // 0/8/16/24

  for (int kb = 0; kb < K; kb += 32) {
    if constexpr (kTdmA) {
      // ---- Tensor Data Mover: DMA the 128x32 bf16 A tile into LDS ----
      if (wave == 0) {
        const unsigned long long gaddr =
            (unsigned long long)A +
            (((unsigned long long)gm0 * (unsigned)lda + (unsigned)kb) << 1);
        v4u g0;
        g0[0] = 1u;                                   // count=1, user descriptor
        g0[1] = (unsigned)(unsigned long long)&As[0][0];  // lds_addr (bytes)
        g0[2] = (unsigned)gaddr;                      // global_addr[31:0]
        g0[3] = (unsigned)(gaddr >> 32) | 0x80000000u;  // addr[56:32] | type=2
        v8u g1;
        g1[0] = (1u << 16)      // data_size = 2 bytes
              | (1u << 20)      // pad_enable
              | (3u << 22)      // pad_interval: every 16 DWORDs (=64B row)
              | (0u << 25);     // pad_amount: 1 DWORD (=4B -> 68B LDS stride)
        g1[1] = ((unsigned)K & 0xFFFFu) << 16;            // tensor_dim0 lo16
        g1[2] = (((unsigned)K >> 16) & 0xFFFFu)           // tensor_dim0 hi16
              | ((T_SEQ & 0xFFFFu) << 16);                // tensor_dim1 lo16
        g1[3] = ((T_SEQ >> 16) & 0xFFFFu)                 // tensor_dim1 hi16
              | (32u << 16);                              // tile_dim0 = 32
        g1[4] = 128u;                                     // tile_dim1 = 128
        g1[5] = (unsigned)lda;                            // dim0_stride lo32
        g1[6] = 0u;
        g1[7] = 0u;
        asm volatile("tensor_load_to_lds %0, %1" :: "s"(g0), "s"(g1) : "memory");
        __builtin_amdgcn_s_wait_tensorcnt(0);
      }
      if (kb + 32 < K)
        __builtin_prefetch((const void*)(A + (size_t)(gm0 + am) * lda + kb + 32), 0, 3);
    } else {
      // ---- VALU staging with fp32->bf16 conversion ----
      const TA* ap = A + (size_t)(gm0 + am) * lda + kb + ak;
#pragma unroll
      for (int j = 0; j < 16; ++j) As[am][ak + j] = ldbf(ap + j);
      if (kb + 32 < K) __builtin_prefetch((const void*)(ap + 32), 0, 3);
    }
    // stage B tile transposed: Bt[k][n] = B[n][k]
    const TB* bp = B + (size_t)(gn0 + bn) * ldb + kb + bk;
#pragma unroll
    for (int j = 0; j < 8; ++j) Bt[bk + j][bn] = ldbf(bp + j);
    if (kb + 32 < K) __builtin_prefetch((const void*)(bp + 32), 0, 3);
    __syncthreads();

    // A fragments (16-bit A layout: lanes 0-15 K={0..7,16..23}, lanes 16-31 +8)
    AFrag a[2], b[2];
#pragma unroll
    for (int i = 0; i < 2; ++i) {
      const int row = wm + i * 16 + lrow;
#pragma unroll
      for (int j = 0; j < 8; ++j) {
        const int kk = (j < 4) ? (kgrp + 2 * j) : (16 + kgrp + 2 * (j - 4));
        a[i].u[j] = *(const unsigned int*)&As[row][kk];
      }
    }
    // B fragments (lane -> N column, packed halfwords -> K pair, same K order as A)
#pragma unroll
    for (int j2 = 0; j2 < 2; ++j2)
#pragma unroll
      for (int j = 0; j < 8; ++j) {
        const int kk = (j < 4) ? (kgrp + 2 * j) : (16 + kgrp + 2 * (j - 4));
        const int bcol = wn + j2 * 16 + lrow;
        b[j2].u[j] = (unsigned int)Bt[kk][bcol] | ((unsigned int)Bt[kk + 1][bcol] << 16);
      }

#pragma unroll
    for (int i = 0; i < 2; ++i)
#pragma unroll
      for (int j2 = 0; j2 < 2; ++j2)
        c[i][j2].v = __builtin_amdgcn_wmma_f32_16x16x32_bf16(
            false, a[i].v, false, b[j2].v, (short)0, c[i][j2].v, false, false);
    __syncthreads();
  }

  // C layout: VGPR r -> M = r + 8*lane[4]; lane[3:0] -> N
#pragma unroll
  for (int i = 0; i < 2; ++i)
#pragma unroll
    for (int j2 = 0; j2 < 2; ++j2)
#pragma unroll
      for (int r = 0; r < 8; ++r) {
        const int row = gm0 + wm + i * 16 + r + (lhi << 3);
        const int col = gn0 + wn + j2 * 16 + lrow;
        stf(&C[(size_t)row * ldc + col], c[i][j2].f[r]);
      }
}

// ---------------------------------------------------------------------------
// Llama-3 RoPE (in place on bf16 [T, nheads*128]), fp32 math
// ---------------------------------------------------------------------------
__global__ __launch_bounds__(256)
void rope_kernel(unsigned short* __restrict__ X, int nheads) {
  const int idx = blockIdx.x * blockDim.x + threadIdx.x;
  const int total = T_SEQ * 64 * nheads;
  if (idx >= total) return;
  const int d = idx & 63;
  const int h = (idx >> 6) % nheads;
  const int t = idx / (64 * nheads);

  float inv = powf(500000.0f, -(float)(2 * d) * (1.0f / 128.0f));
  const float wavelen = 6.283185307179586f / inv;
  float invs = (wavelen > 8192.0f) ? inv * 0.125f : inv;
  const float smooth = (8192.0f / wavelen - 1.0f) * (1.0f / 3.0f);
  const float smoothed = (1.0f - smooth) * 0.125f * inv + smooth * inv;
  if (wavelen <= 8192.0f && wavelen >= 2048.0f) invs = smoothed;

  const float fr = (float)t * invs;
  float sn, cs;
  __sincosf(fr, &sn, &cs);

  const size_t base = (size_t)t * (nheads * HDIM) + h * HDIM + d;
  const float x0 = bf2f(X[base]);
  const float x1 = bf2f(X[base + 64]);
  X[base]      = f2bf(x0 * cs - x1 * sn);
  X[base + 64] = f2bf(x1 * cs + x0 * sn);
}

// ---------------------------------------------------------------------------
// Flash attention (causal, GQA 4:1). One wave per 16-row Q block; 4 waves per
// workgroup share the K/V LDS tiles of one kv head. bf16 WMMA, f32 softmax.
// ---------------------------------------------------------------------------
__global__ __launch_bounds__(128)
void attn_kernel(const unsigned short* __restrict__ Q,
                 const unsigned short* __restrict__ Kb,
                 const unsigned short* __restrict__ Vb,
                 unsigned short* __restrict__ O) {
  const int head = blockIdx.y;
  const int kvh  = head >> 2;
  const int tid  = threadIdx.x;
  const int lane = tid & 31;
  const int wave = tid >> 5;
  const int qr0  = blockIdx.x * 64 + wave * 16;
  const int lrow = lane & 15;
  const int lhi  = lane >> 4;
  const int kgrp = lhi * 8;
  const float scale = 0.08838834764831845f;  // 1/sqrt(128)

  __shared__ unsigned short Kt[128][34];    // [d][s]  (K^T for the S-matmul)
  __shared__ unsigned short Vs[32][130];    // [s][d]
  __shared__ unsigned short Ps[4][16][40];  // per-wave P tile (C->A relayout)

  // Preload Q fragments: 4 x (16x32 bf16) covering d=0..127
  AFrag qf[4];
  {
    const unsigned short* qrow =
        Q + (size_t)(qr0 + lrow) * (NHEADS * HDIM) + head * HDIM;
#pragma unroll
    for (int f = 0; f < 4; ++f)
#pragma unroll
      for (int j = 0; j < 8; ++j) {
        const int d = f * 32 + ((j < 4) ? (kgrp + 2 * j) : (16 + kgrp + 2 * (j - 4)));
        qf[f].u[j] = *(const unsigned int*)(qrow + d);
      }
  }

  CFrag o[8];
  float mrow[8], lsum[8];
#pragma unroll
  for (int n = 0; n < 8; ++n)
#pragma unroll
    for (int r = 0; r < 8; ++r) o[n].f[r] = 0.0f;
#pragma unroll
  for (int r = 0; r < 8; ++r) { mrow[r] = -3.0e38f; lsum[r] = 0.0f; }

  const int send = blockIdx.x * 64 + 64;  // causal upper bound for this WG
  for (int s0 = 0; s0 < send; s0 += 32) {
    // ---- stage K^T and V tiles (shared across the 4 waves) ----
    {
      const int sK = s0 + (tid & 31);
      const int d0 = (tid >> 5) * 32;
      const unsigned short* kp =
          Kb + (size_t)sK * (NKV * HDIM) + kvh * HDIM + d0;
#pragma unroll
      for (int j = 0; j < 32; ++j) Kt[d0 + j][tid & 31] = kp[j];

      const int sv = tid >> 2;
      const int dv = (tid & 3) * 32;
      const unsigned short* vp =
          Vb + (size_t)(s0 + sv) * (NKV * HDIM) + kvh * HDIM + dv;
#pragma unroll
      for (int j = 0; j < 16; ++j)
        *(unsigned int*)&Vs[sv][dv + 2 * j] = *(const unsigned int*)(vp + 2 * j);
      if (s0 + 32 < send) {
        __builtin_prefetch((const void*)(kp + (size_t)32 * (NKV * HDIM)), 0, 3);
        __builtin_prefetch((const void*)(vp + (size_t)32 * (NKV * HDIM)), 0, 3);
      }
    }
    __syncthreads();

    // ---- S = Q * K^T : two 16x16 tiles, reduce over d via 4 WMMAs each ----
    CFrag sf[2];
#pragma unroll
    for (int ct = 0; ct < 2; ++ct) {
#pragma unroll
      for (int r = 0; r < 8; ++r) sf[ct].f[r] = 0.0f;
#pragma unroll
      for (int f = 0; f < 4; ++f) {
        AFrag bk;
        const int scol = ct * 16 + lrow;
#pragma unroll
        for (int j = 0; j < 8; ++j) {
          const int dk = f * 32 + ((j < 4) ? (kgrp + 2 * j) : (16 + kgrp + 2 * (j - 4)));
          bk.u[j] = (unsigned int)Kt[dk][scol] | ((unsigned int)Kt[dk + 1][scol] << 16);
        }
        sf[ct].v = __builtin_amdgcn_wmma_f32_16x16x32_bf16(
            false, qf[f].v, false, bk.v, (short)0, sf[ct].v, false, false);
      }
    }

    // ---- causal mask + online softmax update ----
    float alpha[8];
#pragma unroll
    for (int r = 0; r < 8; ++r) {
      const int grow = qr0 + r + (lhi << 3);
      float v0 = sf[0].f[r] * scale;
      float v1 = sf[1].f[r] * scale;
      if (s0 + lrow > grow)      v0 = -3.0e30f;
      if (s0 + 16 + lrow > grow) v1 = -3.0e30f;
      float mx = fmaxf(v0, v1);
      mx = fmaxf(mx, __shfl_xor(mx, 1, 32));
      mx = fmaxf(mx, __shfl_xor(mx, 2, 32));
      mx = fmaxf(mx, __shfl_xor(mx, 4, 32));
      mx = fmaxf(mx, __shfl_xor(mx, 8, 32));
      const float mn = fmaxf(mrow[r], mx);
      const float al = __expf(mrow[r] - mn);
      const float p0 = __expf(v0 - mn);
      const float p1 = __expf(v1 - mn);
      float rs = p0 + p1;
      rs += __shfl_xor(rs, 1, 32);
      rs += __shfl_xor(rs, 2, 32);
      rs += __shfl_xor(rs, 4, 32);
      rs += __shfl_xor(rs, 8, 32);
      lsum[r] = lsum[r] * al + rs;
      mrow[r] = mn;
      alpha[r] = al;
      Ps[wave][r + (lhi << 3)][lrow]      = f2bf(p0);
      Ps[wave][r + (lhi << 3)][16 + lrow] = f2bf(p1);
    }
#pragma unroll
    for (int n = 0; n < 8; ++n)
#pragma unroll
      for (int r = 0; r < 8; ++r) o[n].f[r] *= alpha[r];
    __syncthreads();  // P visible to all lanes; staging tiles stable

    // ---- O += P * V : 8 WMMAs (one per 16-wide d chunk) ----
    AFrag pf;
#pragma unroll
    for (int j = 0; j < 8; ++j) {
      const int sc = (j < 4) ? (kgrp + 2 * j) : (16 + kgrp + 2 * (j - 4));
      pf.u[j] = *(const unsigned int*)&Ps[wave][lrow][sc];
    }
#pragma unroll
    for (int n = 0; n < 8; ++n) {
      AFrag vf;
#pragma unroll
      for (int j = 0; j < 8; ++j) {
        const int sk = (j < 4) ? (kgrp + 2 * j) : (16 + kgrp + 2 * (j - 4));
        vf.u[j] = (unsigned int)Vs[sk][n * 16 + lrow] | ((unsigned int)Vs[sk + 1][n * 16 + lrow] << 16);
      }
      o[n].v = __builtin_amdgcn_wmma_f32_16x16x32_bf16(
          false, pf.v, false, vf.v, (short)0, o[n].v, false, false);
    }
    __syncthreads();  // done reading Kt/Vs/Ps before next stage
  }

  // ---- epilogue: normalize and store bf16 [T, 32*128] ----
#pragma unroll
  for (int n = 0; n < 8; ++n) {
#pragma unroll
    for (int r = 0; r < 8; ++r) {
      const int row = qr0 + r + (lhi << 3);
      const int col = head * HDIM + n * 16 + lrow;
      O[(size_t)row * (NHEADS * HDIM) + col] = f2bf(o[n].f[r] / lsum[r]);
    }
  }
}

// ---------------------------------------------------------------------------
extern "C" void kernel_launch(void* const* d_in, const int* in_sizes, int n_in,
                              void* d_out, int out_size, void* d_ws, size_t ws_size,
                              hipStream_t stream) {
  const float* x  = (const float*)d_in[0];
  const float* Wq = (const float*)d_in[1];
  const float* Wk = (const float*)d_in[2];
  const float* Wv = (const float*)d_in[3];
  const float* Wo = (const float*)d_in[4];
  float* out = (float*)d_out;

  unsigned short* qb = (unsigned short*)d_ws;                 // [2048, 4096] bf16
  unsigned short* kb = qb + (size_t)T_SEQ * HID;              // [2048, 1024] bf16
  unsigned short* vb = kb + (size_t)T_SEQ * NKV * HDIM;       // [2048, 1024] bf16
  unsigned short* ab = vb + (size_t)T_SEQ * NKV * HDIM;       // [2048, 4096] bf16

  const dim3 blk(256);
  // QKV projections
  gemm_bf16_wmma<float, float, unsigned short>
      <<<dim3(HID / 64, T_SEQ / 128), blk, 0, stream>>>(x, HID, Wq, HID, qb, HID, HID);
  gemm_bf16_wmma<float, float, unsigned short>
      <<<dim3((NKV * HDIM) / 64, T_SEQ / 128), blk, 0, stream>>>(x, HID, Wk, HID, kb, NKV * HDIM, HID);
  gemm_bf16_wmma<float, float, unsigned short>
      <<<dim3((NKV * HDIM) / 64, T_SEQ / 128), blk, 0, stream>>>(x, HID, Wv, HID, vb, NKV * HDIM, HID);
  // RoPE on q and k
  rope_kernel<<<(T_SEQ * NHEADS * 64 + 255) / 256, 256, 0, stream>>>(qb, NHEADS);
  rope_kernel<<<(T_SEQ * NKV * 64 + 255) / 256, 256, 0, stream>>>(kb, NKV);
  // Causal GQA flash attention
  attn_kernel<<<dim3(T_SEQ / 64, NHEADS), 128, 0, stream>>>(qb, kb, vb, ab);
  // Output projection (fp32 out), A tile staged by the Tensor Data Mover
  gemm_bf16_wmma<unsigned short, float, float>
      <<<dim3(HID / 64, T_SEQ / 128), blk, 0, stream>>>(ab, NHEADS * HDIM, Wo, NHEADS * HDIM, out, HID, NHEADS * HDIM);
}
